// TransformerWithCustomAttention_34316788695418
// MI455X (gfx1250) — matrix operand order repaired and checked by reading, not compiled, not run on hardware
//
#include <hip/hip_runtime.h>

typedef __bf16 bf16_t;
typedef __attribute__((ext_vector_type(16))) __bf16 v16bf;
typedef __attribute__((ext_vector_type(8)))  __bf16 v8bf;
typedef __attribute__((ext_vector_type(8)))  float  v8f;
typedef __attribute__((ext_vector_type(4)))  int    v4i;
typedef __attribute__((address_space(1))) v4i* gv4i_p;   // global v4i*
typedef __attribute__((address_space(3))) v4i* lv4i_p;   // LDS v4i*

#define BB 2
#define SS 2048
#define EE 512
#define HH 8
#define LL 4
#define VV 10000
#define DD 64
#define MM (BB*SS)      // 4096 tokens
#define VPAD 10048      // V padded to multiple of 64
#define KT 64           // attention K-tile

// ---------------- async global->LDS helpers (gfx1250 ASYNCcnt path) ---------
static __device__ inline void g2l_b128(const bf16_t* g, bf16_t* l) {
#if __has_builtin(__builtin_amdgcn_global_load_async_to_lds_b128)
    __builtin_amdgcn_global_load_async_to_lds_b128((gv4i_p)(g), (lv4i_p)(l), 0, 0);
#else
    *(v8bf*)l = *(const v8bf*)g;     // sync fallback
#endif
}

static __device__ inline void wait_async0() {
#if __has_builtin(__builtin_amdgcn_s_wait_asynccnt)
    __builtin_amdgcn_s_wait_asynccnt(0);
#else
    asm volatile("s_wait_asynccnt 0x0" ::: "memory");
#endif
}

// ---------------- WMMA fragment helpers (layouts per CDNA5 ISA) ----

// A fragment: 16x32 bf16 (MxK), source row-major with leading dim lda.
// lane L: row = L%16; lanes 0-15 hold K={0..7,16..23}, lanes 16-31 K={8..15,24..31}.
static __device__ inline v16bf load_a_frag(const bf16_t* __restrict__ base, int lda) {
    int lane = threadIdx.x & 31;
    int row  = lane & 15;
    int kb   = (lane >> 4) << 3;      // 0 or 8
    const bf16_t* p = base + (size_t)row * lda + kb;
    v8bf lo = *(const v8bf*)(p);       // K = kb .. kb+7
    v8bf hi = *(const v8bf*)(p + 16);  // K = kb+16 .. kb+23
    return __builtin_shufflevector(lo, hi, 0,1,2,3,4,5,6,7,8,9,10,11,12,13,14,15);
}

// B fragment: 32x16 bf16 (KxN), source row-major [K][N] with leading dim ldb.
// lane L: col = L%16; lanes 0-15 hold K={0..7,16..23}, lanes 16-31 K={8..15,24..31}.
static __device__ inline v16bf load_b_frag(const bf16_t* __restrict__ base, int ldb) {
    int lane = threadIdx.x & 31;
    int n    = lane & 15;
    int kb   = (lane >> 4) << 3;
    v16bf r;
#pragma unroll
    for (int s = 0; s < 8; s++) {
        r[s]     = base[(size_t)(kb + s) * ldb + n];
        r[8 + s] = base[(size_t)(kb + 16 + s) * ldb + n];
    }
    return r;
}

static __device__ inline v8f wmma_bf16(v16bf a, v16bf b, v8f c) {
    return __builtin_amdgcn_wmma_f32_16x16x32_bf16(
        /*neg_a=*/false, a, /*neg_b=*/false, b,
        /*c_mod=*/(short)0, c, /*reuse_a=*/false, /*reuse_b=*/false);
}

// ---------------- weight prep: W[O,I] f32 -> WT[I,Opad] bf16 (transposed) ----
__global__ void transpose_bf16_kernel(const float* __restrict__ W, bf16_t* __restrict__ WT,
                                      int O, int I, int Opad) {
    size_t idx = (size_t)blockIdx.x * 256 + threadIdx.x;
    size_t total = (size_t)I * Opad;
    if (idx >= total) return;
    int o = (int)(idx % Opad);
    int i = (int)(idx / Opad);
    WT[idx] = (o < O) ? (bf16_t)W[(size_t)o * I + i] : (bf16_t)0.0f;
}

// ---------------- embedding + positional encoding --------------------------
__global__ __launch_bounds__(256)
void embed_kernel(const int* __restrict__ tokens, const float* __restrict__ table,
                  const float* __restrict__ pos, float* __restrict__ x,
                  bf16_t* __restrict__ xb) {
    int tkn = blockIdx.x;                 // 0..MM-1
    int s   = tkn % SS;
    int tok = tokens[tkn];
    for (int e = threadIdx.x; e < EE; e += 256) {
        float val = table[(size_t)tok * EE + e] + pos[(size_t)s * EE + e];
        x[(size_t)tkn * EE + e]  = val;
        xb[(size_t)tkn * EE + e] = (bf16_t)val;
    }
}

// ---------------- WMMA GEMM: C[M,N] = A[M,K](bf16) * B[K,N](bf16) + bias ----
// Wave tile 32(M) x 64(N): 2 A-frags reuse each LDS B-frag twice.
// B tile (32 x 64) staged in LDS via async copies, double buffered.
// MODE 0: f32 out, ld=Nreal (guarded)   MODE 2: Q head bf16 [B,H,S,D]
// MODE 3: Kt head bf16 [B,H,D,S]        MODE 4: V head bf16 [B,H,S,D]
template<int MODE>
__global__ __launch_bounds__(256)
void gemm_wmma_kernel(const bf16_t* __restrict__ A, const bf16_t* __restrict__ Bm,
                      const float* __restrict__ bias,
                      float* __restrict__ outF, bf16_t* __restrict__ outH,
                      int M, int Npad, int K, int Nreal) {
    __shared__ __align__(32) bf16_t bsm[2][32 * 64];
    int tid  = threadIdx.x;
    int wave = tid >> 5;
    int lane = tid & 31;
    int m0 = blockIdx.y * 256 + wave * 32;   // 8 waves x 32 rows = 256 rows/block
    int n0 = blockIdx.x * 64;

    // cooperative B-tile copy: 256 threads x 16B = 4KB tile
    int crow = tid >> 3;            // 0..31
    int ccol = (tid & 7) * 8;       // 0..56
    const bf16_t* bsrc = Bm + (size_t)crow * Npad + n0 + ccol;
    bf16_t* bdst0 = &bsm[0][crow * 64 + ccol];
    bf16_t* bdst1 = &bsm[1][crow * 64 + ccol];

    g2l_b128(bsrc, bdst0);          // stage k=0

    v8f c[8] = {};                  // c[t]: rows m0..+15, c[4+t]: rows m0+16..+31
    const bf16_t* arow0 = A + (size_t)m0 * K;
    const bf16_t* arow1 = A + (size_t)(m0 + 16) * K;
    int cur = 0;
    for (int k = 0; k < K; k += 32) {
        wait_async0();
        __syncthreads();
        if (k + 32 < K)
            g2l_b128(bsrc + (size_t)(k + 32) * Npad, cur ? bdst0 : bdst1);
        v16bf a0 = load_a_frag(arow0 + k, K);
        v16bf a1 = load_a_frag(arow1 + k, K);
        const bf16_t* bt = bsm[cur];
#pragma unroll
        for (int t = 0; t < 4; t++) {
            v16bf bf = load_b_frag(bt + t * 16, 64);
            c[t]     = wmma_bf16(a0, bf, c[t]);
            c[4 + t] = wmma_bf16(a1, bf, c[4 + t]);
        }
        __syncthreads();
        cur ^= 1;
    }

    int n  = lane & 15;
    int mh = (lane >> 4) * 8;
#pragma unroll
    for (int half = 0; half < 2; half++) {
#pragma unroll
        for (int t = 0; t < 4; t++) {
            int col = n0 + t * 16 + n;
            if (col >= Nreal) continue;
            float bv = bias ? bias[col] : 0.0f;
#pragma unroll
            for (int vv = 0; vv < 8; vv++) {
                int row = m0 + half * 16 + mh + vv;
                float val = c[half * 4 + t][vv] + bv;
                if (MODE == 0) {
                    outF[(size_t)row * Nreal + col] = val;
                } else {
                    int b = row / SS, s = row % SS;
                    int h = col / DD, d = col % DD;
                    size_t idx;
                    if (MODE == 3)      idx = (((size_t)b * HH + h) * DD + d) * SS + s;
                    else                idx = (((size_t)b * HH + h) * SS + s) * DD + d;
                    outH[idx] = (bf16_t)val;
                }
            }
        }
    }
}

// ---------------- flash attention: softmax(Q Kt / 8) V -> [B,S,E] bf16 ------
// grid (B*H, S/64), block 128 (4 waves); wave owns 16 query rows.
// Kt/V tiles (64x64 each) async-staged to LDS, double buffered.
__global__ __launch_bounds__(128)
void attn_kernel(const bf16_t* __restrict__ Q, const bf16_t* __restrict__ Kt,
                 const bf16_t* __restrict__ V, bf16_t* __restrict__ O) {
    __shared__ __align__(32) bf16_t kt_sm[2][64 * KT];   // [d=64][k=64]
    __shared__ __align__(32) bf16_t v_sm [2][KT * 64];   // [k=64][d=64]
    __shared__ __align__(32) bf16_t p_sm [4][16 * KT];   // per-wave P tile

    int bh   = blockIdx.x;
    int tid  = threadIdx.x;
    int wave = tid >> 5;
    int lane = tid & 31;
    int q0   = blockIdx.y * 64 + wave * 16;

    const bf16_t* Qb  = Q  + ((size_t)bh * SS + q0) * DD;
    const bf16_t* Ktb = Kt + (size_t)bh * DD * SS;
    const bf16_t* Vb  = V  + (size_t)bh * SS * DD;

    v16bf aq0 = load_a_frag(Qb,      DD);   // d = 0..31
    v16bf aq1 = load_a_frag(Qb + 32, DD);   // d = 32..63

    v8f acc[4] = {};
    float Mst[8], Lp[8];
#pragma unroll
    for (int i = 0; i < 8; i++) { Mst[i] = -1e30f; Lp[i] = 0.0f; }

    // cooperative staging: 128 threads, each copies 64B of Kt + 64B of V
    int crow = tid >> 1;            // 0..63
    int cseg = (tid & 1) * 32;      // 0 or 32 (elements)
    auto stage = [&](int kk, int buf) {
        const bf16_t* ks = Ktb + (size_t)crow * SS + kk + cseg;
        const bf16_t* vs = Vb + (size_t)(kk + crow) * DD + cseg;
        bf16_t* kd = &kt_sm[buf][crow * KT + cseg];
        bf16_t* vd = &v_sm[buf][crow * 64 + cseg];
#pragma unroll
        for (int j = 0; j < 4; j++) {
            g2l_b128(ks + j * 8, kd + j * 8);
            g2l_b128(vs + j * 8, vd + j * 8);
        }
    };

    stage(0, 0);
    int cur = 0;
    int n = lane & 15, hi = lane >> 4;
    bf16_t* pbuf = p_sm[wave];

    for (int kk = 0; kk < SS; kk += KT) {
        wait_async0();
        __syncthreads();
        if (kk + KT < SS) stage(kk + KT, cur ^ 1);

        const bf16_t* kt0 = kt_sm[cur];
        const bf16_t* vt0 = v_sm[cur];

        // scores: 16q x 64k tile = 4 sub-tiles, 2 WMMAs each (d chunks)
        v8f s[4];
#pragma unroll
        for (int t = 0; t < 4; t++) {
            v8f z = {};
            z = wmma_bf16(aq0, load_b_frag(kt0 + t * 16,           KT), z);
            z = wmma_bf16(aq1, load_b_frag(kt0 + 32 * KT + t * 16, KT), z);
            s[t] = z;
        }

        // online softmax: only the row-max needs a shuffle tree;
        // row-sum kept as per-lane partials (corr is row-uniform) -> reduce at end.
#pragma unroll
        for (int vv = 0; vv < 8; vv++) {
            float a0 = s[0][vv] * 0.125f;       // 1/sqrt(64)
            float a1 = s[1][vv] * 0.125f;
            float a2 = s[2][vv] * 0.125f;
            float a3 = s[3][vv] * 0.125f;
            float tm = fmaxf(fmaxf(a0, a1), fmaxf(a2, a3));
            tm = fmaxf(tm, __shfl_xor(tm, 1));
            tm = fmaxf(tm, __shfl_xor(tm, 2));
            tm = fmaxf(tm, __shfl_xor(tm, 4));
            tm = fmaxf(tm, __shfl_xor(tm, 8));
            float newM = fmaxf(Mst[vv], tm);
            float corr = __expf(Mst[vv] - newM);
            Mst[vv] = newM;
            float e0 = __expf(a0 - newM);
            float e1 = __expf(a1 - newM);
            float e2 = __expf(a2 - newM);
            float e3 = __expf(a3 - newM);
            Lp[vv] = Lp[vv] * corr + (e0 + e1 + e2 + e3);
            acc[0][vv] *= corr; acc[1][vv] *= corr;
            acc[2][vv] *= corr; acc[3][vv] *= corr;
            int row = vv + 8 * hi;
            pbuf[row * KT + n]      = (bf16_t)e0;
            pbuf[row * KT + 16 + n] = (bf16_t)e1;
            pbuf[row * KT + 32 + n] = (bf16_t)e2;
            pbuf[row * KT + 48 + n] = (bf16_t)e3;
        }
        asm volatile("" ::: "memory");  // keep LDS store->load order (same-wave DS in-order)
        v16bf pf0 = load_a_frag(pbuf,      KT);   // k-local 0..31
        v16bf pf1 = load_a_frag(pbuf + 32, KT);   // k-local 32..63

#pragma unroll
        for (int t = 0; t < 4; t++) {
            acc[t] = wmma_bf16(pf0, load_b_frag(vt0 + t * 16,           64), acc[t]);
            acc[t] = wmma_bf16(pf1, load_b_frag(vt0 + 32 * 64 + t * 16, 64), acc[t]);
        }
        __syncthreads();
        cur ^= 1;
    }

    // finish deferred row-sum reduction, then normalized epilogue
#pragma unroll
    for (int vv = 0; vv < 8; vv++) {
        float ts = Lp[vv];
        ts += __shfl_xor(ts, 1);
        ts += __shfl_xor(ts, 2);
        ts += __shfl_xor(ts, 4);
        ts += __shfl_xor(ts, 8);
        Lp[vv] = 1.0f / ts;
    }
    int b = bh / HH, h = bh % HH;
#pragma unroll
    for (int t = 0; t < 4; t++)
#pragma unroll
        for (int vv = 0; vv < 8; vv++) {
            int row = q0 + vv + 8 * hi;
            int col = h * DD + t * 16 + n;
            O[((size_t)b * SS + row) * EE + col] = (bf16_t)(acc[t][vv] * Lp[vv]);
        }
}

// ---------------- residual add + LayerNorm -> x (f32) and xb (bf16) ---------
__global__ __launch_bounds__(256)
void add_ln_kernel(float* __restrict__ x, const float* __restrict__ a,
                   const float* __restrict__ g, const float* __restrict__ bta,
                   bf16_t* __restrict__ xb) {
    int tkn = blockIdx.x;
    int tid = threadIdx.x;
    float* xr = x + (size_t)tkn * EE;
    const float* ar = a + (size_t)tkn * EE;
    float v0 = xr[tid]       + ar[tid];
    float v1 = xr[tid + 256] + ar[tid + 256];
    __shared__ float s1[256], s2[256];
    s1[tid] = v0 + v1;
    s2[tid] = v0 * v0 + v1 * v1;
    __syncthreads();
    for (int off = 128; off > 0; off >>= 1) {
        if (tid < off) { s1[tid] += s1[tid + off]; s2[tid] += s2[tid + off]; }
        __syncthreads();
    }
    float mu  = s1[0] * (1.0f / EE);
    float var = s2[0] * (1.0f / EE) - mu * mu;
    float rs  = rsqrtf(var + 1e-5f);
    float o0 = (v0 - mu) * rs * g[tid]       + bta[tid];
    float o1 = (v1 - mu) * rs * g[tid + 256] + bta[tid + 256];
    xr[tid]       = o0;
    xr[tid + 256] = o1;
    xb[(size_t)tkn * EE + tid]       = (bf16_t)o0;
    xb[(size_t)tkn * EE + tid + 256] = (bf16_t)o1;
}

// ---------------- host driver ----------------------------------------------
extern "C" void kernel_launch(void* const* d_in, const int* in_sizes, int n_in,
                              void* d_out, int out_size, void* d_ws, size_t ws_size,
                              hipStream_t stream) {
    const int*   tokens = (const int*)  d_in[0];
    const float* table  = (const float*)d_in[1];
    const float* pos    = (const float*)d_in[2];
    const float* Wq     = (const float*)d_in[3];
    const float* bq     = (const float*)d_in[4];
    const float* Wk     = (const float*)d_in[5];
    const float* bk     = (const float*)d_in[6];
    const float* Wv     = (const float*)d_in[7];
    const float* bv     = (const float*)d_in[8];
    const float* Wo     = (const float*)d_in[9];
    const float* bo     = (const float*)d_in[10];
    const float* lng    = (const float*)d_in[11];
    const float* lnb    = (const float*)d_in[12];
    const float* Wout   = (const float*)d_in[13];
    const float* bout   = (const float*)d_in[14];

    // carve workspace (256B aligned slabs)
    size_t off = 0;
    auto carve = [&](size_t bytes) -> void* {
        void* p = (char*)d_ws + off;
        off += (bytes + 255) & ~(size_t)255;
        return p;
    };
    float*  x     = (float*) carve((size_t)MM * EE * 4);
    bf16_t* xb    = (bf16_t*)carve((size_t)MM * EE * 2);
    bf16_t* qbuf  = (bf16_t*)carve((size_t)MM * EE * 2);
    bf16_t* ktbuf = (bf16_t*)carve((size_t)MM * EE * 2);
    bf16_t* vbuf  = (bf16_t*)carve((size_t)MM * EE * 2);
    bf16_t* attnb = (bf16_t*)carve((size_t)MM * EE * 2);
    float*  aout  = (float*) carve((size_t)MM * EE * 4);
    bf16_t* wT    = (bf16_t*)carve((size_t)LL * 4 * EE * EE * 2); // Wq,Wk,Wv,Wo ^T per layer
    bf16_t* woutT = (bf16_t*)carve((size_t)EE * VPAD * 2);

    const size_t WSZ = (size_t)EE * EE;
    dim3 tgrid((unsigned)((WSZ + 255) / 256));
    for (int l = 0; l < LL; l++) {
        transpose_bf16_kernel<<<tgrid, 256, 0, stream>>>(Wq + l * WSZ, wT + (4 * l + 0) * WSZ, EE, EE, EE);
        transpose_bf16_kernel<<<tgrid, 256, 0, stream>>>(Wk + l * WSZ, wT + (4 * l + 1) * WSZ, EE, EE, EE);
        transpose_bf16_kernel<<<tgrid, 256, 0, stream>>>(Wv + l * WSZ, wT + (4 * l + 2) * WSZ, EE, EE, EE);
        transpose_bf16_kernel<<<tgrid, 256, 0, stream>>>(Wo + l * WSZ, wT + (4 * l + 3) * WSZ, EE, EE, EE);
    }
    {
        size_t total = (size_t)EE * VPAD;
        transpose_bf16_kernel<<<dim3((unsigned)((total + 255) / 256)), 256, 0, stream>>>(
            Wout, woutT, VV, EE, VPAD);
    }

    embed_kernel<<<MM, 256, 0, stream>>>(tokens, table, pos, x, xb);

    dim3 ggrid(EE / 64, MM / 256);      // (8, 16)
    for (int l = 0; l < LL; l++) {
        gemm_wmma_kernel<2><<<ggrid, 256, 0, stream>>>(xb, wT + (4 * l + 0) * WSZ, bq + l * EE,
                                                       nullptr, qbuf, MM, EE, EE, EE);
        gemm_wmma_kernel<3><<<ggrid, 256, 0, stream>>>(xb, wT + (4 * l + 1) * WSZ, bk + l * EE,
                                                       nullptr, ktbuf, MM, EE, EE, EE);
        gemm_wmma_kernel<4><<<ggrid, 256, 0, stream>>>(xb, wT + (4 * l + 2) * WSZ, bv + l * EE,
                                                       nullptr, vbuf, MM, EE, EE, EE);
        attn_kernel<<<dim3(BB * HH, SS / 64), 128, 0, stream>>>(qbuf, ktbuf, vbuf, attnb);
        gemm_wmma_kernel<0><<<ggrid, 256, 0, stream>>>(attnb, wT + (4 * l + 3) * WSZ, bo + l * EE,
                                                       aout, nullptr, MM, EE, EE, EE);
        add_ln_kernel<<<MM, 256, 0, stream>>>(x, aout, lng + l * EE, lnb + l * EE, xb);
    }

    // final vocab projection straight into d_out (f32, ld = 10000, N-tail guarded)
    gemm_wmma_kernel<0><<<dim3(VPAD / 64, MM / 256), 256, 0, stream>>>(
        xb, woutT, bout, (float*)d_out, nullptr, MM, VPAD, EE, VV);
}
